// VMambaBlock_1726576854848
// MI455X (gfx1250) — matrix operand order repaired and checked by reading, not compiled, not run on hardware
//
#include <hip/hip_runtime.h>
#include <hip/hip_bf16.h>

typedef __attribute__((ext_vector_type(16))) __bf16 v16bf;
typedef __attribute__((ext_vector_type(8)))  __bf16 v8bf;
typedef __attribute__((ext_vector_type(8)))  float  v8f;

constexpr int kB   = 2;
constexpr int kL   = 1024;
constexpr int kDM  = 1024;
constexpr int kED  = 2048;
constexpr int kN   = 16;
constexpr int kDTR = 64;
constexpr int kDC  = 4;
constexpr int kBL  = kB * kL;
constexpr int kDBC = kDTR + 2 * kN;   // 96

// ---------------- elementwise conversion kernels ----------------

__global__ void k_cvt(const float* __restrict__ in, __bf16* __restrict__ out, int n) {
  int i = blockIdx.x * blockDim.x + threadIdx.x;
  if (i < n) out[i] = (__bf16)in[i];
}

// in: (R,C) fp32 row-major  ->  out: (C,R) bf16 row-major (i.e. transposed)
__global__ void k_cvt_t(const float* __restrict__ in, __bf16* __restrict__ out, int R, int C) {
  int i = blockIdx.x * blockDim.x + threadIdx.x;
  if (i < R * C) {
    int r = i / C, c = i % C;
    out[(size_t)c * R + r] = (__bf16)in[i];
  }
}

// dbc (rows,96) fp32 -> delta part (rows,64) bf16
__global__ void k_extract(const float* __restrict__ dbc, __bf16* __restrict__ out, int rows) {
  int i = blockIdx.x * blockDim.x + threadIdx.x;
  if (i < rows * kDTR) {
    int r = i / kDTR, c = i % kDTR;
    out[i] = (__bf16)dbc[(size_t)r * kDBC + c];
  }
}

// ---------------- depthwise causal conv1d + SiLU ----------------
// xz: (BL, 2*ED); xc half is columns [0,ED). Left pad kDC-1.
__global__ void k_conv(const float* __restrict__ xz, const float* __restrict__ w,
                       const float* __restrict__ bia, float* __restrict__ xcf,
                       __bf16* __restrict__ xcbf) {
  int i = blockIdx.x * blockDim.x + threadIdx.x;
  if (i >= kB * kL * kED) return;
  int e = i % kED;
  int l = (i / kED) % kL;
  int b = i / (kED * kL);
  float acc = bia[e];
#pragma unroll
  for (int k = 0; k < kDC; ++k) {
    int ls = l - (kDC - 1) + k;
    if (ls >= 0)
      acc += xz[(size_t)(b * kL + ls) * (2 * kED) + e] * w[e * kDC + k];
  }
  float s = acc / (1.0f + __expf(-acc));   // SiLU
  xcf[i]  = s;
  xcbf[i] = (__bf16)s;
}

// ---------------- bf16 WMMA GEMM: C(M,N) = A(M,K) @ B(K,N) ----------------
// A: row-major bf16 (M,K).  Bt: B transposed, row-major bf16 (N,K).  C fp32 (M,N).
// Each wave computes a 16 x (16*NT) strip. Per K-step: A fragment + ALL NT B
// fragments are loaded into distinct registers first (one load clause), then NT
// back-to-back WMMAs with independent accumulators run behind staggered partial
// loadcnt waits, overlapping matrix math with the remaining loads.
template <int NT>
__global__ __launch_bounds__(256) void k_gemm_bf16(const __bf16* __restrict__ A,
                                                   const __bf16* __restrict__ Bt,
                                                   float* __restrict__ C,
                                                   int M, int N, int K) {
  const int lane = threadIdx.x & 31;
  const int wave = threadIdx.x >> 5;
  const int tiles_m  = M >> 4;
  const int groups_n = N / (16 * NT);
  int g = blockIdx.x * (blockDim.x >> 5) + wave;
  if (g >= tiles_m * groups_n) return;         // wave-uniform: EXEC all-ones inside
  const int tm = g % tiles_m;
  const int gn = g / tiles_m;
  const int half = lane >> 4;                  // 0: lanes 0-15, 1: lanes 16-31
  const int l16  = lane & 15;
  // A fragment: row M = tm*16+l16; lane-half selects K sub-ranges {0..7,16..23} / {8..15,24..31}
  const int klo = half * 8;
  // B fragment: col N = tile*16+l16; lane-half selects K sub-ranges {0..7,16..23} / {8..15,24..31}
  const int kha = half * 16;

  const __bf16* arow = A + (size_t)(tm * 16 + l16) * K;
  const __bf16* brow[NT];
#pragma unroll
  for (int j = 0; j < NT; ++j)
    brow[j] = Bt + (size_t)((gn * NT + j) * 16 + l16) * K;

  v8f c[NT];
#pragma unroll
  for (int j = 0; j < NT; ++j) c[j] = (v8f){};

  union U { v16bf v; v8bf h[2]; };

  for (int kb = 0; kb < K; kb += 32) {
    U a;
    a.h[0] = *(const v8bf*)(arow + kb + klo);
    a.h[1] = *(const v8bf*)(arow + kb + 16 + klo);
    U b[NT];                       // distinct registers per tile -> staggered waits
#pragma unroll
    for (int j = 0; j < NT; ++j) {
      b[j].h[0] = *(const v8bf*)(brow[j] + kb + klo);
      b[j].h[1] = *(const v8bf*)(brow[j] + kb + 16 + klo);
    }
    if (kb + 256 < K) __builtin_prefetch(arow + kb + 256, 0, 1);
#pragma unroll
    for (int j = 0; j < NT; ++j)
      c[j] = __builtin_amdgcn_wmma_f32_16x16x32_bf16(false, a.v, false, b[j].v,
                                                     (short)0, c[j], false, false);
  }
  // C/D layout: VGPR r -> M = tm*16 + r + half*8, N = tile*16 + l16
#pragma unroll
  for (int j = 0; j < NT; ++j) {
    float* crow = C + (size_t)(tm * 16 + half * 8) * N + (gn * NT + j) * 16 + l16;
#pragma unroll
    for (int r = 0; r < 8; ++r) crow[(size_t)r * N] = c[j][r];
  }
}

// ---------------- fused selective scan + gate ----------------
// 16 lanes = the N states of one (b,e) channel; wave32 holds 2 channels.
// h_t = exp(delta*A) * h_{t-1} + delta*B_t*x_t ; y = sum_n h*C + D*x ; y *= silu(z)
__global__ __launch_bounds__(256) void k_scan(const float* __restrict__ dRaw,  // (BL,ED)
                                              const float* __restrict__ dtb,   // (ED)
                                              const float* __restrict__ A_log, // (ED,N)
                                              const float* __restrict__ Dvec,  // (ED)
                                              const float* __restrict__ xcf,   // (BL,ED)
                                              const float* __restrict__ dbc,   // (BL,96)
                                              const float* __restrict__ xz,    // (BL,2ED)
                                              __bf16* __restrict__ ybf) {      // (BL,ED)
  const int tid = blockIdx.x * blockDim.x + threadIdx.x;
  const int p = tid >> 4;          // (b,e) channel index
  const int n = tid & 15;          // state index
  if (p >= kB * kED) return;
  const int b = p / kED;
  const int e = p % kED;

  const float Aen  = -__expf(A_log[e * kN + n]);
  const float bias = dtb[e];
  const float Dv   = Dvec[e];
  float h = 0.0f;

  size_t oE = (size_t)b * kL * kED + e;                 // per-l stride kED
  size_t oD = (size_t)b * kL * kDBC;                    // per-l stride 96
  size_t oZ = (size_t)b * kL * (2 * kED) + kED + e;     // z column; stride 2*ED

  for (int l = 0; l < kL; ++l) {
    float draw = dRaw[oE] + bias;
    float d    = (draw > 20.0f) ? draw : log1pf(__expf(draw));  // softplus
    float xv   = xcf[oE];
    float Bm   = dbc[oD + kDTR + n];
    float Cm   = dbc[oD + kDTR + kN + n];
    float dA   = __expf(d * Aen);
    h = dA * h + d * Bm * xv;
    float acc = h * Cm;
    // reduce over the 16 states (xor masks < 16 stay inside the group on wave32)
    acc += __shfl_xor(acc, 8, 32);
    acc += __shfl_xor(acc, 4, 32);
    acc += __shfl_xor(acc, 2, 32);
    acc += __shfl_xor(acc, 1, 32);
    if (n == 0) {
      float z = xz[oZ];
      float y = (acc + Dv * xv) * (z / (1.0f + __expf(-z)));
      ybf[oE] = (__bf16)y;
    }
    oE += kED; oD += kDBC; oZ += 2 * kED;
  }
}

// ---------------- host launcher ----------------

extern "C" void kernel_launch(void* const* d_in, const int* in_sizes, int n_in,
                              void* d_out, int out_size, void* d_ws, size_t ws_size,
                              hipStream_t stream) {
  (void)in_sizes; (void)n_in; (void)out_size; (void)ws_size;
  const float* x       = (const float*)d_in[0];  // (B,L,DM)
  const float* in_w    = (const float*)d_in[1];  // (DM, 2ED)
  const float* conv_w  = (const float*)d_in[2];  // (ED,1,4)
  const float* conv_b  = (const float*)d_in[3];  // (ED)
  const float* xproj_w = (const float*)d_in[4];  // (ED, 96)
  const float* dt_w    = (const float*)d_in[5];  // (64, ED)
  const float* dt_b    = (const float*)d_in[6];  // (ED)
  const float* A_log   = (const float*)d_in[7];  // (ED, N)
  const float* Dv      = (const float*)d_in[8];  // (ED)
  const float* out_w   = (const float*)d_in[9];  // (ED, DM)
  float* out = (float*)d_out;                    // (B,L,DM)

  char* base = (char*)d_ws;
  size_t o = 0;
  auto alloc = [&](size_t bytes) -> char* {
    char* r = base + o;
    o = (o + bytes + 255) & ~(size_t)255;
    return r;
  };

  float*  xz   = (float*) alloc((size_t)kBL * 2 * kED * 4);
  float*  xcf  = (float*) alloc((size_t)kBL * kED * 4);
  float*  dbc  = (float*) alloc((size_t)kBL * kDBC * 4);
  float*  dRaw = (float*) alloc((size_t)kBL * kED * 4);
  __bf16* xbf  = (__bf16*)alloc((size_t)kBL * kDM * 2);
  __bf16* w1t  = (__bf16*)alloc((size_t)2 * kED * kDM * 2);
  __bf16* xcbf = (__bf16*)alloc((size_t)kBL * kED * 2);
  __bf16* w2t  = (__bf16*)alloc((size_t)kDBC * kED * 2);
  __bf16* dprt = (__bf16*)alloc((size_t)kBL * kDTR * 2);
  __bf16* wdtt = (__bf16*)alloc((size_t)kED * kDTR * 2);
  __bf16* ybf  = (__bf16*)alloc((size_t)kBL * kED * 2);
  __bf16* w3t  = (__bf16*)alloc((size_t)kDM * kED * 2);

  const int T = 256;

  // bf16 conversions (weights transposed so WMMA B-fragments are contiguous loads)
  k_cvt  <<<(kBL * kDM + T - 1) / T, T, 0, stream>>>(x, xbf, kBL * kDM);
  k_cvt_t<<<(kDM * 2 * kED + T - 1) / T, T, 0, stream>>>(in_w,    w1t,  kDM, 2 * kED);
  k_cvt_t<<<(kED * kDBC + T - 1) / T, T, 0, stream>>>(xproj_w, w2t,  kED, kDBC);
  k_cvt_t<<<(kDTR * kED + T - 1) / T, T, 0, stream>>>(dt_w,    wdtt, kDTR, kED);
  k_cvt_t<<<(kED * kDM + T - 1) / T, T, 0, stream>>>(out_w,   w3t,  kED, kDM);

  auto gemm = [&](const __bf16* A, const __bf16* Bt, float* C, int M, int N, int K) {
    int wpb = T / 32;
    if (N % 64 == 0) {
      int groups = (M / 16) * (N / 64);
      k_gemm_bf16<4><<<(groups + wpb - 1) / wpb, T, 0, stream>>>(A, Bt, C, M, N, K);
    } else {
      int groups = (M / 16) * (N / 32);
      k_gemm_bf16<2><<<(groups + wpb - 1) / wpb, T, 0, stream>>>(A, Bt, C, M, N, K);
    }
  };

  gemm(xbf, w1t, xz, kBL, 2 * kED, kDM);                               // in_proj
  k_conv<<<(kB * kL * kED + T - 1) / T, T, 0, stream>>>(xz, conv_w, conv_b, xcf, xcbf);
  gemm(xcbf, w2t, dbc, kBL, kDBC, kED);                                // x_proj (N=96 -> NT=2)
  k_extract<<<(kBL * kDTR + T - 1) / T, T, 0, stream>>>(dbc, dprt, kBL);
  gemm(dprt, wdtt, dRaw, kBL, kED, kDTR);                              // dt_proj
  k_scan<<<(kB * kED * 16 + T - 1) / T, T, 0, stream>>>(dRaw, dt_b, A_log, Dv,
                                                        xcf, dbc, xz, ybf);
  gemm(ybf, w3t, out, kBL, kDM, kED);                                  // out_proj
}
